// WSGraphCL_31361851195743
// MI455X (gfx1250) — matrix operand order repaired and checked by reading, not compiled, not run on hardware
//
#include <hip/hip_runtime.h>
#include <hip/hip_bf16.h>
#include <math.h>

// ---------------------------------------------------------------------------
// WSGraphCL GCN pipeline for MI455X (gfx1250, wave32, WMMA).
//
// All GEMMs run as NT bf16 WMMA (v_wmma_f32_16x16x32_bf16, f32 accum).
// A_hat in bf16 is 75 MB -> resident in the 192 MB L2 across all 6 layers,
// so the dominant 116 GFLOP of A_hat@(HW) is WMMA-compute-bound, not HBM-bound.
// ---------------------------------------------------------------------------

#define N_NODES 6144
#define C_IN    200
#define C_PAD   256
#define DM_     256
#define DEPTH_  6
#define PH_     512
#define PD_     256
#define KNN     10
#define ETA_    0.5f
#define DELTA_  1.0f
#define BN_EPS_ 1e-5f

typedef __bf16 bf16_t;
typedef __bf16 v8bf  __attribute__((ext_vector_type(8)));
typedef __bf16 v16bf __attribute__((ext_vector_type(16)));
typedef float  v8f   __attribute__((ext_vector_type(8)));

// ---------------------------------------------------------------------------
// NT GEMM: C[M x N] = A[M x K] * B[N x K]^T  (+ optional bias over N)
// bf16 inputs, f32 accumulate via v_wmma_f32_16x16x32_bf16.
// Block = 256 threads = 8 wave32; block tile 128x128; wave tile 32x64.
// Requires: M % 128 == 0, N % 128 == 0, K % 32 == 0 (guaranteed by padding).
// ---------------------------------------------------------------------------
__global__ __launch_bounds__(256) void gemm_nt_bf16_kernel(
    const bf16_t* __restrict__ A, const bf16_t* __restrict__ B,
    float* __restrict__ C, int M, int Ncols, int K,
    const float* __restrict__ bias)
{
  const int wid     = threadIdx.x >> 5;   // 0..7
  const int lane    = threadIdx.x & 31;
  const int laneLow = lane & 15;
  const int laneHi  = lane >> 4;          // 0 or 1

  const int m0 = blockIdx.y * 128 + (wid >> 1) * 32;  // 4 wave-rows of 32
  const int n0 = blockIdx.x * 128 + (wid & 1)  * 64;  // 2 wave-cols of 64

  const v8f vz = {0.f, 0.f, 0.f, 0.f, 0.f, 0.f, 0.f, 0.f};
  v8f acc[2][4];
#pragma unroll
  for (int t = 0; t < 2; t++)
#pragma unroll
    for (int u = 0; u < 4; u++) acc[t][u] = vz;

  const size_t rowA0 = (size_t)(m0 + laneLow) * (size_t)K;
  const size_t rowB0 = (size_t)(n0 + laneLow) * (size_t)K;

  for (int k = 0; k < K; k += 32) {
    // A fragment, 16x32 bf16 per 16-row subtile:
    //   lanes 0-15 : row m, K = k+{0..7} then k+{16..23}
    //   lanes 16-31: row m, K = k+{8..15} then k+{24..31}
    v16bf afrag[2];
#pragma unroll
    for (int t = 0; t < 2; t++) {
      const bf16_t* p = A + rowA0 + (size_t)t * 16 * K + k + laneHi * 8;
      v8bf lo = *(const v8bf*)(p);
      v8bf hi = *(const v8bf*)(p + 16);
      afrag[t] = __builtin_shufflevector(lo, hi,
          0, 1, 2, 3, 4, 5, 6, 7, 8, 9, 10, 11, 12, 13, 14, 15);
    }
    // B fragment, 32x16 bf16 per 16-col subtile (rows striped across lanes):
    //   lane n (0-15): col n, K = k+{0..7} then k+{16..23} ; lane n+16: col n, K = k+{8..15} then k+{24..31}
    v16bf bfrag[4];
#pragma unroll
    for (int u = 0; u < 4; u++) {
      const bf16_t* p = B + rowB0 + (size_t)u * 16 * K + k + laneHi * 8;
      v8bf lo = *(const v8bf*)(p);
      v8bf hi = *(const v8bf*)(p + 16);
      bfrag[u] = __builtin_shufflevector(lo, hi,
          0, 1, 2, 3, 4, 5, 6, 7, 8, 9, 10, 11, 12, 13, 14, 15);
    }

    if (k + 32 < K) {
      // Locality 3 -> WGP-scope prefetch, filling every cache level on a miss
      // (SYS-scope prefetch would skip the WGP cache entirely).
      __builtin_prefetch(A + rowA0 + k + 64, 0, 3);
      __builtin_prefetch(B + rowB0 + k + 64, 0, 3);
    }

#pragma unroll
    for (int t = 0; t < 2; t++)
#pragma unroll
      for (int u = 0; u < 4; u++)
        acc[t][u] = __builtin_amdgcn_wmma_f32_16x16x32_bf16(
            false, afrag[t], false, bfrag[u],
            (short)0, acc[t][u], false, false);
  }

  // C/D layout: VGPR r -> (M = r + 8*laneHi, N = laneLow) within each 16x16.
#pragma unroll
  for (int t = 0; t < 2; t++)
#pragma unroll
    for (int u = 0; u < 4; u++) {
      const int n  = n0 + u * 16 + laneLow;
      const float bv = bias ? bias[n] : 0.0f;
#pragma unroll
      for (int r = 0; r < 8; r++) {
        const int m = m0 + t * 16 + laneHi * 8 + r;
        C[(size_t)m * (size_t)Ncols + n] = acc[t][u][r] + bv;
      }
    }
}

// ---------------------------------------------------------------------------
// Elementwise / reduction helper kernels (bandwidth-trivial at 23.3 TB/s)
// ---------------------------------------------------------------------------
__global__ void fill_kernel(float* __restrict__ p, long long count, float v) {
  for (long long i = (long long)blockIdx.x * blockDim.x + threadIdx.x;
       i < count; i += (long long)gridDim.x * blockDim.x)
    p[i] = v;
}

__global__ void pad_cvt_kernel(const float* __restrict__ in, bf16_t* __restrict__ out,
                               int rows, int cols, int colsP) {
  long long total = (long long)rows * colsP;
  for (long long i = (long long)blockIdx.x * blockDim.x + threadIdx.x;
       i < total; i += (long long)gridDim.x * blockDim.x) {
    int r = (int)(i / colsP), c = (int)(i % colsP);
    out[i] = (c < cols) ? (bf16_t)in[(size_t)r * cols + c] : (bf16_t)0.f;
  }
}

// out[c*Rp + r] = (r < R) ? in[r*C + c] : 0   (f32 -> bf16, zero-pad K dim)
__global__ void transpose_cvt_kernel(const float* __restrict__ in, bf16_t* __restrict__ out,
                                     int R, int Ccols, int Rp) {
  long long total = (long long)Ccols * Rp;
  for (long long i = (long long)blockIdx.x * blockDim.x + threadIdx.x;
       i < total; i += (long long)gridDim.x * blockDim.x) {
    int c = (int)(i / Rp), r = (int)(i % Rp);
    out[i] = (r < R) ? (bf16_t)in[(size_t)r * Ccols + c] : (bf16_t)0.f;
  }
}

__global__ void cvt_bf16_kernel(const float* __restrict__ in, bf16_t* __restrict__ out,
                                long long count) {
  for (long long i = (long long)blockIdx.x * blockDim.x + threadIdx.x;
       i < count; i += (long long)gridDim.x * blockDim.x)
    out[i] = (bf16_t)in[i];
}

__global__ void rowsumsq_kernel(const float* __restrict__ X, float* __restrict__ a2,
                                int rows, int cols) {
  const int i = blockIdx.x;
  float s = 0.f;
  for (int c = threadIdx.x; c < cols; c += blockDim.x) {
    float v = X[(size_t)i * cols + c];
    s += v * v;
  }
  __shared__ float sm[256];
  sm[threadIdx.x] = s;
  __syncthreads();
  for (int st = 128; st > 0; st >>= 1) {
    if ((int)threadIdx.x < st) sm[threadIdx.x] += sm[threadIdx.x + st];
    __syncthreads();
  }
  if (threadIdx.x == 0) a2[i] = sm[0];
}

// Pass 1: G (Gram) -> Dx in place; track global max(Dp), max(Dx) via bit-atomics.
__global__ void dist_pass1_kernel(float* __restrict__ G, const float* __restrict__ a2,
                                  const float* __restrict__ P, unsigned* __restrict__ mx,
                                  int n) {
  float lmP = 0.f, lmX = 0.f;
  long long total = (long long)n * n;
  for (long long idx = (long long)blockIdx.x * blockDim.x + threadIdx.x;
       idx < total; idx += (long long)gridDim.x * blockDim.x) {
    int i = (int)(idx / n), j = (int)(idx % n);
    float d2 = a2[i] + a2[j] - 2.0f * G[idx];
    float dx = (d2 > 0.f) ? sqrtf(d2) : 0.f;   // clip + safe_sqrt
    G[idx] = dx;
    float dr = P[2 * i] - P[2 * j], dc = P[2 * i + 1] - P[2 * j + 1];
    float pp = dr * dr + dc * dc;
    float dp = (pp > 0.f) ? sqrtf(pp) : 0.f;
    lmP = fmaxf(lmP, dp);
    lmX = fmaxf(lmX, dx);
  }
  __shared__ float sp[256], sx[256];
  sp[threadIdx.x] = lmP; sx[threadIdx.x] = lmX;
  __syncthreads();
  for (int st = 128; st > 0; st >>= 1) {
    if ((int)threadIdx.x < st) {
      sp[threadIdx.x] = fmaxf(sp[threadIdx.x], sp[threadIdx.x + st]);
      sx[threadIdx.x] = fmaxf(sx[threadIdx.x], sx[threadIdx.x + st]);
    }
    __syncthreads();
  }
  if (threadIdx.x == 0) {  // non-negative floats: uint order == float order
    atomicMax(mx + 0, __float_as_uint(sp[0]));
    atomicMax(mx + 1, __float_as_uint(sx[0]));
  }
}

// Pass 2: D = ETA * norm(Dp) + (1-ETA) * norm(Dx), in place over G.
__global__ void dist_pass2_kernel(float* __restrict__ D, const float* __restrict__ P,
                                  const unsigned* __restrict__ mx, int n) {
  const float mp = __uint_as_float(mx[0]);
  const float mv = __uint_as_float(mx[1]);
  long long total = (long long)n * n;
  for (long long idx = (long long)blockIdx.x * blockDim.x + threadIdx.x;
       idx < total; idx += (long long)gridDim.x * blockDim.x) {
    int i = (int)(idx / n), j = (int)(idx % n);
    float dx = D[idx];
    float dr = P[2 * i] - P[2 * j], dc = P[2 * i + 1] - P[2 * j + 1];
    float pp = dr * dr + dc * dc;
    float dp = (pp > 0.f) ? sqrtf(pp) : 0.f;
    float np_ = (mp > 0.f) ? dp / (mp + 1e-8f) : dp;
    float nx_ = (mv > 0.f) ? dx / (mv + 1e-8f) : dx;
    D[idx] = ETA_ * np_ + (1.0f - ETA_) * nx_;
  }
}

// Per-row top-K nearest (excluding self): 256-thread partial top-10 lists,
// log-tree merge in LDS, then mark A[i][j] = A[j][i] = 1 (races benign: same value).
__global__ void topk_adj_kernel(const float* __restrict__ D, float* __restrict__ A, int n) {
  const int i = blockIdx.x;
  const int tid = threadIdx.x;
  float bd[KNN]; int bi[KNN];
#pragma unroll
  for (int t = 0; t < KNN; t++) { bd[t] = __builtin_inff(); bi[t] = -1; }
  for (int j = tid; j < n; j += 256) {
    float d = (j == i) ? __builtin_inff() : D[(size_t)i * n + j];
    if (d < bd[KNN - 1]) {
      int p = KNN - 1;
      while (p > 0 && bd[p - 1] > d) { bd[p] = bd[p - 1]; bi[p] = bi[p - 1]; p--; }
      bd[p] = d; bi[p] = j;
    }
  }
  __shared__ float sd[256][KNN];
  __shared__ int   si[256][KNN];
#pragma unroll
  for (int t = 0; t < KNN; t++) { sd[tid][t] = bd[t]; si[tid][t] = bi[t]; }
  __syncthreads();
  for (int s = 128; s > 0; s >>= 1) {
    if (tid < s) {
      float md[KNN]; int mi[KNN];
      int p = 0, q = 0;
#pragma unroll
      for (int t = 0; t < KNN; t++) {
        float d1 = sd[tid][p], d2 = sd[tid + s][q];
        if (d1 <= d2) { md[t] = d1; mi[t] = si[tid][p]; p++; }
        else          { md[t] = d2; mi[t] = si[tid + s][q]; q++; }
      }
#pragma unroll
      for (int t = 0; t < KNN; t++) { sd[tid][t] = md[t]; si[tid][t] = mi[t]; }
    }
    __syncthreads();
  }
  if (tid < KNN) {
    int j = si[0][tid];
    if (j >= 0) {
      A[(size_t)i * n + j] = 1.0f;
      A[(size_t)j * n + i] = 1.0f;
    }
  }
}

// A_w = where(A>0, exp(-D^2/(delta^2+1e-8)), 0) + I   (in place on A)
__global__ void heat_adj_kernel(float* __restrict__ Aw, const float* __restrict__ D, int n) {
  long long total = (long long)n * n;
  for (long long idx = (long long)blockIdx.x * blockDim.x + threadIdx.x;
       idx < total; idx += (long long)gridDim.x * blockDim.x) {
    int i = (int)(idx / n), j = (int)(idx % n);
    float d = D[idx];
    float w = (Aw[idx] > 0.f) ? expf(-d * d / (DELTA_ * DELTA_ + 1e-8f)) : 0.f;
    Aw[idx] = w + ((i == j) ? 1.0f : 0.0f);
  }
}

// dis[i] = rsqrt(max(sum_j Aw[i][j], 1))
__global__ void degree_kernel(const float* __restrict__ Aw, float* __restrict__ dis, int n) {
  const int i = blockIdx.x;
  float s = 0.f;
  for (int j = threadIdx.x; j < n; j += 256) s += Aw[(size_t)i * n + j];
  __shared__ float sm[256];
  sm[threadIdx.x] = s;
  __syncthreads();
  for (int st = 128; st > 0; st >>= 1) {
    if ((int)threadIdx.x < st) sm[threadIdx.x] += sm[threadIdx.x + st];
    __syncthreads();
  }
  if (threadIdx.x == 0) dis[i] = rsqrtf(fmaxf(sm[0], 1.0f));
}

// A_hat (bf16) = dis[i] * dis[j] * Aw[i][j]
__global__ void norm_cvt_kernel(const float* __restrict__ Aw, const float* __restrict__ dis,
                                bf16_t* __restrict__ Ahb, int n) {
  long long total = (long long)n * n;
  for (long long idx = (long long)blockIdx.x * blockDim.x + threadIdx.x;
       idx < total; idx += (long long)gridDim.x * blockDim.x) {
    int i = (int)(idx / n), j = (int)(idx % n);
    Ahb[idx] = (bf16_t)(dis[i] * dis[j] * Aw[idx]);
  }
}

// Per-column mean / biased variance over `rows` rows.
__global__ void colstats_kernel(const float* __restrict__ M, int rows, int cols,
                                float* __restrict__ mu, float* __restrict__ var) {
  const int j = blockIdx.x;
  float s = 0.f, s2 = 0.f;
  for (int i = threadIdx.x; i < rows; i += 256) {
    float v = M[(size_t)i * cols + j];
    s += v; s2 += v * v;
  }
  __shared__ float ss[256], ss2[256];
  ss[threadIdx.x] = s; ss2[threadIdx.x] = s2;
  __syncthreads();
  for (int st = 128; st > 0; st >>= 1) {
    if ((int)threadIdx.x < st) {
      ss[threadIdx.x]  += ss[threadIdx.x + st];
      ss2[threadIdx.x] += ss2[threadIdx.x + st];
    }
    __syncthreads();
  }
  if (threadIdx.x == 0) {
    float m = ss[0] / (float)rows;
    mu[j]  = m;
    var[j] = ss2[0] / (float)rows - m * m;
  }
}

// out_bf16 = PReLU(BN(M))  with per-column gamma/beta and scalar slope a[a_idx].
__global__ void bn_prelu_cvt_kernel(const float* __restrict__ M, long long count, int cols,
                                    const float* __restrict__ mu, const float* __restrict__ var,
                                    const float* __restrict__ g, const float* __restrict__ b,
                                    int gb_off, const float* __restrict__ a, int a_idx,
                                    bf16_t* __restrict__ out) {
  const float slope = a[a_idx];
  for (long long idx = (long long)blockIdx.x * blockDim.x + threadIdx.x;
       idx < count; idx += (long long)gridDim.x * blockDim.x) {
    int j = (int)(idx % cols);
    float v = (M[idx] - mu[j]) * rsqrtf(var[j] + BN_EPS_) * g[gb_off + j] + b[gb_off + j];
    v = (v >= 0.f) ? v : slope * v;
    out[idx] = (bf16_t)v;
  }
}

// ---------------------------------------------------------------------------
static inline char* bump(char*& p, size_t bytes) {
  char* r = p;
  p += (bytes + 255) & ~(size_t)255;
  return r;
}

extern "C" void kernel_launch(void* const* d_in, const int* in_sizes, int n_in,
                              void* d_out, int out_size, void* d_ws, size_t ws_size,
                              hipStream_t stream) {
  (void)in_sizes; (void)n_in; (void)out_size; (void)ws_size;

  const float* X       = (const float*)d_in[0];   // [N, C]
  const float* P       = (const float*)d_in[1];   // [N, 2]
  const float* W0      = (const float*)d_in[2];   // [C, DM]
  const float* Wl      = (const float*)d_in[3];   // [DEPTH-1, DM, DM]
  const float* gammas  = (const float*)d_in[4];   // [DEPTH, DM]
  const float* betas   = (const float*)d_in[5];   // [DEPTH, DM]
  const float* prelu_a = (const float*)d_in[6];   // [DEPTH]
  const float* P1      = (const float*)d_in[7];   // [DM, PH]
  const float* g1      = (const float*)d_in[8];   // [PH]
  const float* b1      = (const float*)d_in[9];   // [PH]
  const float* a_p     = (const float*)d_in[10];  // [1]
  const float* P2      = (const float*)d_in[11];  // [PH, PD]
  const float* b2      = (const float*)d_in[12];  // [PD]
  float* out = (float*)d_out;                     // [N, PD]

  const int n = N_NODES;
  char* wp = (char*)d_ws;
  float*   G    = (float*)bump(wp, (size_t)n * n * 4);            // Gram -> Dx -> D
  float*   Aw   = (float*)bump(wp, (size_t)n * n * 4);            // adjacency -> A_w
  bf16_t*  Ahb  = (bf16_t*)bump(wp, (size_t)n * n * 2);           // A_hat bf16 (L2-resident)
  bf16_t*  Xb   = (bf16_t*)bump(wp, (size_t)n * C_PAD * 2);
  float*   a2   = (float*)bump(wp, (size_t)n * 4);
  float*   dis  = (float*)bump(wp, (size_t)n * 4);
  unsigned* mx  = (unsigned*)bump(wp, 256);
  bf16_t*  Wt   = (bf16_t*)bump(wp, (size_t)DEPTH_ * DM_ * C_PAD * 2); // W^T per layer
  bf16_t*  P1t  = (bf16_t*)bump(wp, (size_t)PH_ * DM_ * 2);
  bf16_t*  P2t  = (bf16_t*)bump(wp, (size_t)PD_ * PH_ * 2);
  bf16_t*  Hb   = (bf16_t*)bump(wp, (size_t)n * DM_ * 2);
  float*   HWt  = (float*)bump(wp, (size_t)DM_ * n * 4);
  bf16_t*  HWtb = (bf16_t*)bump(wp, (size_t)DM_ * n * 2);
  float*   Mb   = (float*)bump(wp, (size_t)n * DM_ * 4);
  float*   U    = (float*)bump(wp, (size_t)n * PH_ * 4);
  bf16_t*  Zb   = (bf16_t*)bump(wp, (size_t)n * PH_ * 2);
  float*   mu   = (float*)bump(wp, (size_t)PH_ * 4);
  float*   var  = (float*)bump(wp, (size_t)PH_ * 4);

  const dim3 blk(256);
  const int EW = 4096;  // grid for grid-stride elementwise kernels

  // --- graph construction -------------------------------------------------
  fill_kernel<<<EW, blk, 0, stream>>>(Aw, (long long)n * n, 0.0f);
  fill_kernel<<<1, blk, 0, stream>>>((float*)mx, 2, 0.0f);
  pad_cvt_kernel<<<EW, blk, 0, stream>>>(X, Xb, n, C_IN, C_PAD);
  rowsumsq_kernel<<<n, blk, 0, stream>>>(X, a2, n, C_IN);

  // Pre-transpose all weights to NT-GEMM layout (bf16, K zero-padded).
  transpose_cvt_kernel<<<256, blk, 0, stream>>>(W0, Wt, C_IN, DM_, C_PAD);
  for (int l = 0; l < DEPTH_ - 1; l++)
    transpose_cvt_kernel<<<256, blk, 0, stream>>>(
        Wl + (size_t)l * DM_ * DM_, Wt + (size_t)(l + 1) * DM_ * C_PAD, DM_, DM_, DM_);
  transpose_cvt_kernel<<<512, blk, 0, stream>>>(P1, P1t, DM_, PH_, DM_);
  transpose_cvt_kernel<<<512, blk, 0, stream>>>(P2, P2t, PH_, PD_, PH_);

  // Gram matrix X X^T (NT: A = B = Xb), then combined normalized distance D.
  gemm_nt_bf16_kernel<<<dim3(n / 128, n / 128), blk, 0, stream>>>(
      Xb, Xb, G, n, n, C_PAD, nullptr);
  dist_pass1_kernel<<<EW, blk, 0, stream>>>(G, a2, P, mx, n);
  dist_pass2_kernel<<<EW, blk, 0, stream>>>(G, P, mx, n);

  // kNN adjacency, heat-kernel weights + I, symmetric normalization -> bf16.
  topk_adj_kernel<<<n, blk, 0, stream>>>(G, Aw, n);
  heat_adj_kernel<<<EW, blk, 0, stream>>>(Aw, G, n);
  degree_kernel<<<n, blk, 0, stream>>>(Aw, dis, n);
  norm_cvt_kernel<<<EW, blk, 0, stream>>>(Aw, dis, Ahb, n);

  // --- GCN layers ---------------------------------------------------------
  // (H@W)^T = W^T @ H^T  => gemm_nt(Wt, Hb) directly yields HWt [DM x N],
  // which is exactly the B operand layout needed by gemm_nt(Ahb, HWtb).
  for (int l = 0; l < DEPTH_; l++) {
    const bf16_t* Hin = (l == 0) ? Xb : Hb;
    gemm_nt_bf16_kernel<<<dim3(n / 128, DM_ / 128), blk, 0, stream>>>(
        Wt + (size_t)l * DM_ * C_PAD, Hin, HWt, DM_, n, C_PAD, nullptr);
    cvt_bf16_kernel<<<EW, blk, 0, stream>>>(HWt, HWtb, (long long)DM_ * n);
    gemm_nt_bf16_kernel<<<dim3(DM_ / 128, n / 128), blk, 0, stream>>>(
        Ahb, HWtb, Mb, n, DM_, n, nullptr);
    colstats_kernel<<<DM_, blk, 0, stream>>>(Mb, n, DM_, mu, var);
    bn_prelu_cvt_kernel<<<EW, blk, 0, stream>>>(
        Mb, (long long)n * DM_, DM_, mu, var, gammas, betas, l * DM_, prelu_a, l, Hb);
  }

  // --- projector + output -------------------------------------------------
  gemm_nt_bf16_kernel<<<dim3(PH_ / 128, n / 128), blk, 0, stream>>>(
      Hb, P1t, U, n, PH_, DM_, nullptr);
  colstats_kernel<<<PH_, blk, 0, stream>>>(U, n, PH_, mu, var);
  bn_prelu_cvt_kernel<<<EW, blk, 0, stream>>>(
      U, (long long)n * PH_, PH_, mu, var, g1, b1, 0, a_p, 0, Zb);
  gemm_nt_bf16_kernel<<<dim3(PD_ / 128, n / 128), blk, 0, stream>>>(
      Zb, P2t, out, n, PD_, PH_, b2);
}
